// A2LM_LMAudioPrev_35931696398849
// MI455X (gfx1250) — matrix operand order repaired and checked by reading, not compiled, not run on hardware
//
#include <hip/hip_runtime.h>
#include <hip/hip_bf16.h>

#define B_   64
#define T_   200
#define D_   336
#define DPAD 352           // 11 K-tiles of 32
#define H_   512
#define NBLK 8             // 8 WGPs x 32 waves = 256 waves: 1 wave per tile-job,
#define NTHR 1024          // 8 waves/SIMD for L2-latency hiding (WGP max)

typedef __attribute__((ext_vector_type(16))) __bf16       bf16x16;
typedef __attribute__((ext_vector_type(8)))  float        f32x8;
typedef __attribute__((ext_vector_type(4)))  unsigned int u32x4;

union FragU { u32x4 u[2]; bf16x16 v; };

// ---------------- global-address-space forced accessors ----------------
typedef __attribute__((address_space(1))) const u32x4*   gu32x4cp;
typedef __attribute__((address_space(1))) const float*   gfloatcp;
typedef __attribute__((address_space(1))) float*         gfloatp;
typedef __attribute__((address_space(1))) __bf16*        gbf16p;

__device__ __forceinline__ bf16x16 gload32B(const __bf16* p) {
  gu32x4cp g = (gu32x4cp)(unsigned long long)p;
  FragU f;
  f.u[0] = g[0];
  f.u[1] = g[1];
  return f.v;
}
__device__ __forceinline__ float gloadf(const float* p) {
  return *(gfloatcp)(unsigned long long)p;
}
__device__ __forceinline__ void gstoref(float* p, float v) {
  *(gfloatp)(unsigned long long)p = v;
}
__device__ __forceinline__ void gstoreh(__bf16* p, __bf16 v) {
  *(gbf16p)(unsigned long long)p = v;
}

__device__ __forceinline__ float sigf(float x) { return 1.0f / (1.0f + __expf(-x)); }
__device__ __forceinline__ float tanhfast(float x) {
  float e = __expf(2.0f * x);
  return (e - 1.0f) / (e + 1.0f);
}

#define WMMA_BF16(a, b, c) \
  __builtin_amdgcn_wmma_f32_16x16x32_bf16(false, (a), false, (b), (short)0, (c), false, false)

// ---------------- params ----------------
struct Params {
  const __bf16* xbf;       // [T][B][DPAD] bf16
  const __bf16* w[6];      // packed cell weights (B-operand tiles)
  const float*  bias[6];   // [2048] each
  const __bf16* wfc1; const float* bfc1;
  const __bf16* wfc2; const float* bfc2;
  __bf16* hbuf;            // [6 cells][2 parity][B*H] bf16
  float*  cbuf;            // [6][B*H] fp32
  __bf16* fc1out;          // [B*256] bf16
  float*  out;             // [B][T][136] fp32
  unsigned* barCnt;
  unsigned* barSense;
  int nblocks;
};

__device__ __forceinline__ __bf16* Hbuf(const Params& p, int cell, int par) {
  return p.hbuf + (size_t)(cell * 2 + par) * (B_ * H_);
}

// ---------------- grid barrier ----------------
__device__ void gridBarrier(unsigned* cnt, unsigned* sense, int nblocks) {
  __threadfence();
  __syncthreads();
  if (threadIdx.x == 0) {
    unsigned gen = __hip_atomic_load(sense, __ATOMIC_ACQUIRE, __HIP_MEMORY_SCOPE_AGENT);
    unsigned a = __hip_atomic_fetch_add(cnt, 1u, __ATOMIC_ACQ_REL, __HIP_MEMORY_SCOPE_AGENT);
    if (a == (unsigned)nblocks - 1u) {
      __hip_atomic_store(cnt, 0u, __ATOMIC_RELAXED, __HIP_MEMORY_SCOPE_AGENT);
      __hip_atomic_fetch_add(sense, 1u, __ATOMIC_RELEASE, __HIP_MEMORY_SCOPE_AGENT);
    } else {
      while (__hip_atomic_load(sense, __ATOMIC_ACQUIRE, __HIP_MEMORY_SCOPE_AGENT) == gen)
        __builtin_amdgcn_s_sleep(2);
    }
  }
  __syncthreads();
}

// ---------------- one LSTM-cell wave job (CELL is compile-time) ----------------
// Computes a (16 batch x 16 h) tile of all 4 gates, then the cell update.
template <int CELL>
__device__ void cell_job(const Params& p, int mTile, int hTile,
                         int lane, int t, int pp, int pc) {
  constexpr int kT = (CELL == 0) ? 27 : (CELL == 3) ? 43 : 32;
  constexpr size_t GOFS = (size_t)32 * kT * 512;   // elements between gate blocks

  const __bf16* xt = p.xbf + (size_t)t * B_ * DPAD;
  int mBase = mTile * 16;
  int aLaneOff = (lane >> 4) << 4;   // +16 elements for upper half-wave
  int mRow = mBase + (lane & 15);

  f32x8 acc0 = {0.f,0.f,0.f,0.f,0.f,0.f,0.f,0.f};
  f32x8 acc1 = acc0, acc2 = acc0, acc3 = acc0;

  const __bf16* wp = p.w[CELL] + (size_t)hTile * kT * 512 + lane * 16;

  auto run_seg = [&](const __bf16* sp, int stride, int tiles) {
    const __bf16* ap = sp + (size_t)mRow * stride + aLaneOff;
    for (int i = 0; i < tiles; ++i) {
      bf16x16 a  = gload32B(ap);
      bf16x16 b0 = gload32B(wp);
      bf16x16 b1 = gload32B(wp + GOFS);
      bf16x16 b2 = gload32B(wp + 2 * GOFS);
      bf16x16 b3 = gload32B(wp + 3 * GOFS);
      acc0 = WMMA_BF16(a, b0, acc0);
      acc1 = WMMA_BF16(a, b1, acc1);
      acc2 = WMMA_BF16(a, b2, acc2);
      acc3 = WMMA_BF16(a, b3, acc3);
      ap += 32;
      wp += 512;
    }
  };

  if (CELL == 0) {                 // audio L0: x then own h
    run_seg(xt, DPAD, 11);
    run_seg(Hbuf(p, 0, pp), H_, 16);
  } else if (CELL == 1) {
    run_seg(Hbuf(p, 0, pc), H_, 16);
    run_seg(Hbuf(p, 1, pp), H_, 16);
  } else if (CELL == 2) {
    run_seg(Hbuf(p, 1, pc), H_, 16);
    run_seg(Hbuf(p, 2, pp), H_, 16);
  } else if (CELL == 3) {          // lm L0: aprev (= audio h3), x, own h
    run_seg(Hbuf(p, 2, pp), H_, 16);
    run_seg(xt, DPAD, 11);
    run_seg(Hbuf(p, 3, pp), H_, 16);
  } else if (CELL == 4) {
    run_seg(Hbuf(p, 3, pc), H_, 16);
    run_seg(Hbuf(p, 4, pp), H_, 16);
  } else {
    run_seg(Hbuf(p, 4, pc), H_, 16);
    run_seg(Hbuf(p, 5, pp), H_, 16);
  }

  // in-register LSTM cell update; this wave exclusively owns these (b,h) pairs
  int n = hTile * 16 + (lane & 15);
  const float* bias = p.bias[CELL];
  float bi  = gloadf(bias + n);
  float bff = gloadf(bias + 512 + n);
  float bg  = gloadf(bias + 1024 + n);
  float bo  = gloadf(bias + 1536 + n);
  float* cst = p.cbuf + (size_t)CELL * B_ * H_;
  __bf16* hout = Hbuf(p, CELL, pc);
  int rowBase = mBase + ((lane >> 4) << 3);
#pragma unroll
  for (int r = 0; r < 8; ++r) {
    size_t off = (size_t)(rowBase + r) * H_ + n;
    float cprev = gloadf(cst + off);
    float iv = sigf(acc0[r] + bi);
    float fv = sigf(acc1[r] + bff);
    float gv = tanhfast(acc2[r] + bg);
    float ov = sigf(acc3[r] + bo);
    float cnew = fv * cprev + iv * gv;
    gstoref(cst + off, cnew);
    gstoreh(hout + off, (__bf16)(ov * tanhfast(cnew)));
  }
}

// ---------------- FC head jobs (pipelined one step behind) ----------------
__device__ void fc1_job(const Params& p, int job, int lane, int pHist) {
  int mTile = job & 3, nTile = job >> 2;   // nTile 0..15
  const __bf16* hin = Hbuf(p, 5, pHist);
  f32x8 acc = {0.f,0.f,0.f,0.f,0.f,0.f,0.f,0.f};
  const __bf16* ap = hin + (size_t)(mTile * 16 + (lane & 15)) * H_ + ((lane >> 4) << 4);
  const __bf16* wpc = p.wfc1 + (size_t)nTile * 16 * 512 + lane * 16;
  for (int kt = 0; kt < 16; ++kt) {
    bf16x16 a = gload32B(ap);
    bf16x16 b = gload32B(wpc);
    acc = WMMA_BF16(a, b, acc);
    ap += 32;
    wpc += 512;
  }
  int n = nTile * 16 + (lane & 15);
  float bb = gloadf(p.bfc1 + n);
  int rowBase = mTile * 16 + ((lane >> 4) << 3);
#pragma unroll
  for (int r = 0; r < 8; ++r)
    gstoreh(p.fc1out + (size_t)(rowBase + r) * 256 + n, (__bf16)(acc[r] + bb));
}

__device__ void fc2_job(const Params& p, int job, int lane, int tOut) {
  int mTile = job & 3, nTile = job >> 2;   // nTile 0..8 (144 padded cols)
  f32x8 acc = {0.f,0.f,0.f,0.f,0.f,0.f,0.f,0.f};
  const __bf16* ap = p.fc1out + (size_t)(mTile * 16 + (lane & 15)) * 256 + ((lane >> 4) << 4);
  const __bf16* wpc = p.wfc2 + (size_t)nTile * 8 * 512 + lane * 16;
  for (int kt = 0; kt < 8; ++kt) {
    bf16x16 a = gload32B(ap);
    bf16x16 b = gload32B(wpc);
    acc = WMMA_BF16(a, b, acc);
    ap += 32;
    wpc += 512;
  }
  int n = nTile * 16 + (lane & 15);
  if (n < 136) {
    float bb = gloadf(p.bfc2 + n);
    int rowBase = mTile * 16 + ((lane >> 4) << 3);
#pragma unroll
    for (int r = 0; r < 8; ++r)
      gstoref(p.out + ((size_t)(rowBase + r) * T_ + tOut) * 136 + n, acc[r] + bb);
  }
}

// ---------------- persistent scan kernel ----------------
__global__ __launch_bounds__(NTHR, 1) void lstm_persistent(Params p) {
  int lane = threadIdx.x & 31;
  int wave = blockIdx.x * (blockDim.x >> 5) + (threadIdx.x >> 5);
  int nwaves = gridDim.x * (blockDim.x >> 5);
  int pp = 0;

  // prelude: audio_step on frame 0 (no_grad prelude in reference)
  for (int j = wave; j < 128; j += nwaves)
    cell_job<0>(p, (j >> 5) & 3, j & 31, lane, 0, pp, pp ^ 1);
  gridBarrier(p.barCnt, p.barSense, p.nblocks);
  for (int j = wave; j < 128; j += nwaves)
    cell_job<1>(p, (j >> 5) & 3, j & 31, lane, 0, pp, pp ^ 1);
  gridBarrier(p.barCnt, p.barSense, p.nblocks);
  for (int j = wave; j < 128; j += nwaves)
    cell_job<2>(p, (j >> 5) & 3, j & 31, lane, 0, pp, pp ^ 1);
  gridBarrier(p.barCnt, p.barSense, p.nblocks);
  pp ^= 1;

  for (int t = 0; t < T_; ++t) {
    int pc = pp ^ 1;
    // P1: lm0 + audio0 + FC1(t-1)
    int nj = (t > 0) ? 320 : 256;
    for (int j = wave; j < nj; j += nwaves) {
      if (j < 128)      cell_job<3>(p, (j >> 5) & 3, j & 31, lane, t, pp, pc);
      else if (j < 256) cell_job<0>(p, ((j - 128) >> 5) & 3, (j - 128) & 31, lane, t, pp, pc);
      else              fc1_job(p, j - 256, lane, pp);
    }
    gridBarrier(p.barCnt, p.barSense, p.nblocks);
    // P2: lm1 + audio1 + FC2(t-1)
    nj = (t > 0) ? 292 : 256;
    for (int j = wave; j < nj; j += nwaves) {
      if (j < 128)      cell_job<4>(p, (j >> 5) & 3, j & 31, lane, t, pp, pc);
      else if (j < 256) cell_job<1>(p, ((j - 128) >> 5) & 3, (j - 128) & 31, lane, t, pp, pc);
      else              fc2_job(p, j - 256, lane, t - 1);
    }
    gridBarrier(p.barCnt, p.barSense, p.nblocks);
    // P3: lm2 + audio2
    for (int j = wave; j < 256; j += nwaves) {
      if (j < 128) cell_job<5>(p, (j >> 5) & 3, j & 31, lane, t, pp, pc);
      else         cell_job<2>(p, ((j - 128) >> 5) & 3, (j - 128) & 31, lane, t, pp, pc);
    }
    gridBarrier(p.barCnt, p.barSense, p.nblocks);
    pp = pc;
  }
  // epilogue: FC head for t = T-1
  for (int j = wave; j < 64; j += nwaves) fc1_job(p, j, lane, pp);
  gridBarrier(p.barCnt, p.barSense, p.nblocks);
  for (int j = wave; j < 36; j += nwaves) fc2_job(p, j, lane, T_ - 1);
}

// ---------------- prep kernels ----------------
// Pack W = [W_ih | pad | W_hh] (row-major [N, K]) into WMMA-B tiles:
// tile(nTile,kTile) is 32 lanes x 16 bf16; lane l -> col n=l&15, K half=(l>>4)*16+j.
__global__ void pack_weights(const float* Wih, const float* Whh, __bf16* dst,
                             int N, int Kih, int KihPad, int Khh, int total) {
  int idx = blockIdx.x * blockDim.x + threadIdx.x;
  if (idx >= total) return;
  int kTiles = (KihPad + Khh) >> 5;
  int j = idx & 15;
  int lane = (idx >> 4) & 31;
  int tile = idx >> 9;
  int kTile = tile % kTiles;
  int nTile = tile / kTiles;
  int n = nTile * 16 + (lane & 15);
  int k = kTile * 32 + ((lane >> 4) << 4) + j;
  float v = 0.f;
  if (n < N) {
    if (k < KihPad) { if (k < Kih) v = Wih[(size_t)n * Kih + k]; }
    else            { v = Whh[(size_t)n * Khh + (k - KihPad)]; }
  }
  dst[idx] = (__bf16)v;
}

__global__ void convert_x(const float* audio, __bf16* xbf, int total) {
  int idx = blockIdx.x * blockDim.x + threadIdx.x;
  if (idx >= total) return;
  int k = idx % DPAD;
  int r = idx / DPAD;
  int b = r % B_;
  int t = r / B_;
  float v = (k < D_) ? audio[((size_t)b * T_ + t) * D_ + k] : 0.f;
  xbf[idx] = (__bf16)v;
}

__global__ void init_state(const float* h0a, const float* c0a,
                           const float* h0l, const float* c0l,
                           __bf16* hbuf, float* cbuf) {
  int idx = blockIdx.x * blockDim.x + threadIdx.x;
  if (idx >= B_ * H_) return;
  __bf16 ha = (__bf16)h0a[idx];
  __bf16 hl = (__bf16)h0l[idx];
  for (int c = 0; c < 3; ++c) {
    hbuf[(size_t)(c * 2 + 0) * B_ * H_ + idx] = ha;
    hbuf[(size_t)(c * 2 + 1) * B_ * H_ + idx] = ha;
    cbuf[(size_t)c * B_ * H_ + idx] = c0a[idx];
    hbuf[(size_t)((c + 3) * 2 + 0) * B_ * H_ + idx] = hl;
    hbuf[(size_t)((c + 3) * 2 + 1) * B_ * H_ + idx] = hl;
    cbuf[(size_t)(c + 3) * B_ * H_ + idx] = c0l[idx];
  }
}

// ---------------- host entry ----------------
extern "C" void kernel_launch(void* const* d_in, const int* in_sizes, int n_in,
                              void* d_out, int out_size, void* d_ws, size_t ws_size,
                              hipStream_t stream) {
  (void)in_sizes; (void)n_in; (void)out_size; (void)ws_size;
  const float* audio  = (const float*)d_in[0];
  const float* h0a    = (const float*)d_in[1];
  const float* c0a    = (const float*)d_in[2];
  const float* h0l    = (const float*)d_in[3];
  const float* c0l    = (const float*)d_in[4];
  const float* aW_ih0 = (const float*)d_in[5];
  const float* aW_hh0 = (const float*)d_in[6];
  const float* ab0    = (const float*)d_in[7];
  const float* aW_ih  = (const float*)d_in[8];
  const float* aW_hh  = (const float*)d_in[9];
  const float* ab     = (const float*)d_in[10];
  const float* lW_ih0 = (const float*)d_in[11];
  const float* lW_hh0 = (const float*)d_in[12];
  const float* lb0    = (const float*)d_in[13];
  const float* lW_ih  = (const float*)d_in[14];
  const float* lW_hh  = (const float*)d_in[15];
  const float* lb     = (const float*)d_in[16];
  const float* fc1_w  = (const float*)d_in[17];
  const float* fc1_b  = (const float*)d_in[18];
  const float* fc2_w  = (const float*)d_in[19];
  const float* fc2_b  = (const float*)d_in[20];

  // ---- carve workspace ----
  size_t off = 0;
  auto carve = [&](size_t bytes) -> void* {
    void* ptr = (char*)d_ws + off;
    off = (off + bytes + 255) & ~(size_t)255;
    return ptr;
  };
  unsigned* bar    = (unsigned*)carve(256);
  __bf16* xbf      = (__bf16*)carve((size_t)T_ * B_ * DPAD * 2);
  __bf16* wa0      = (__bf16*)carve((size_t)128 * 27 * 512 * 2);
  __bf16* wa1      = (__bf16*)carve((size_t)128 * 32 * 512 * 2);
  __bf16* wa2      = (__bf16*)carve((size_t)128 * 32 * 512 * 2);
  __bf16* wl0      = (__bf16*)carve((size_t)128 * 43 * 512 * 2);
  __bf16* wl1      = (__bf16*)carve((size_t)128 * 32 * 512 * 2);
  __bf16* wl2      = (__bf16*)carve((size_t)128 * 32 * 512 * 2);
  __bf16* wfc1     = (__bf16*)carve((size_t)16 * 16 * 512 * 2);
  __bf16* wfc2     = (__bf16*)carve((size_t)9 * 8 * 512 * 2);
  __bf16* hbuf     = (__bf16*)carve((size_t)6 * 2 * B_ * H_ * 2);
  float*  cbuf     = (float*)carve((size_t)6 * B_ * H_ * 4);
  __bf16* fc1out   = (__bf16*)carve((size_t)B_ * 256 * 2);

  (void)hipMemsetAsync(bar, 0, 256, stream);

  auto pack = [&](const float* Wih, const float* Whh, __bf16* dst,
                  int N, int Npad, int Kih, int KihPad, int Khh) {
    int total = (Npad / 16) * ((KihPad + Khh) / 32) * 512;
    pack_weights<<<(total + 255) / 256, 256, 0, stream>>>(Wih, Whh, dst, N, Kih, KihPad, Khh, total);
  };
  const int NW = 2048, HW = 512;
  pack(aW_ih0, aW_hh0, wa0, 2048, 2048, 336, DPAD, 512);
  pack(aW_ih + 0 * (size_t)NW * HW, aW_hh + 0 * (size_t)NW * HW, wa1, 2048, 2048, 512, 512, 512);
  pack(aW_ih + 1 * (size_t)NW * HW, aW_hh + 1 * (size_t)NW * HW, wa2, 2048, 2048, 512, 512, 512);
  pack(lW_ih0, lW_hh0, wl0, 2048, 2048, 848, 864, 512);
  pack(lW_ih + 0 * (size_t)NW * HW, lW_hh + 0 * (size_t)NW * HW, wl1, 2048, 2048, 512, 512, 512);
  pack(lW_ih + 1 * (size_t)NW * HW, lW_hh + 1 * (size_t)NW * HW, wl2, 2048, 2048, 512, 512, 512);
  pack(fc1_w, nullptr, wfc1, 256, 256, 512, 512, 0);
  pack(fc2_w, nullptr, wfc2, 136, 144, 256, 256, 0);

  {
    int total = T_ * B_ * DPAD;
    convert_x<<<(total + 255) / 256, 256, 0, stream>>>(audio, xbf, total);
  }
  init_state<<<(B_ * H_ + 255) / 256, 256, 0, stream>>>(h0a, c0a, h0l, c0l, hbuf, cbuf);

  Params p;
  p.xbf = xbf;
  p.w[0] = wa0; p.w[1] = wa1; p.w[2] = wa2;
  p.w[3] = wl0; p.w[4] = wl1; p.w[5] = wl2;
  p.bias[0] = ab0; p.bias[1] = ab; p.bias[2] = ab + 2048;
  p.bias[3] = lb0; p.bias[4] = lb; p.bias[5] = lb + 2048;
  p.wfc1 = wfc1; p.bfc1 = fc1_b;
  p.wfc2 = wfc2; p.bfc2 = fc2_b;
  p.hbuf = hbuf; p.cbuf = cbuf; p.fc1out = fc1out;
  p.out = (float*)d_out;
  p.barCnt = bar; p.barSense = bar + 16;
  p.nblocks = NBLK;

  lstm_persistent<<<NBLK, NTHR, 0, stream>>>(p);
}
